// RNN_lxg_13417477833140
// MI455X (gfx1250) — matrix operand order repaired and checked by reading, not compiled, not run on hardware
//
#include <hip/hip_runtime.h>
#include <hip/hip_bf16.h>
#include <cstdint>
#include <cstddef>

// ---------------------------------------------------------------------------
// Problem constants (from reference): B=16, L=2048, NIN=NHID=1024, 2 layers.
// Each layer (h_prev==0) = GEMM  g[M,3H] = x[M,K] @ W[3H,K]^T  + gate math.
// M = 32768, K = 1024, H = 1024.  Compute-bound => bf16 WMMA with hi/lo split
// (3 products) for ~fp32 accuracy; all splits done OUTSIDE the GEMM loop.
// ---------------------------------------------------------------------------
#define GRU_B   16
#define GRU_L   2048
#define GRU_K   1024
#define GRU_H   1024
#define GRU_M   (GRU_B * GRU_L)        // 32768

typedef __bf16 bf16_t;
typedef __attribute__((ext_vector_type(16))) __bf16 v16bf;
typedef __attribute__((ext_vector_type(8)))  __bf16 v8bf;
typedef __attribute__((ext_vector_type(8)))  float  v8f;

__device__ __forceinline__ float sigmoid_f(float x) {
    return 1.0f / (1.0f + __expf(-x));
}

// ---------------------------------------------------------------------------
// Split fp32 -> bf16 hi + bf16 lo planes (hi = rn(x), lo = rn(x - hi)).
// Used for both weight matrices and the layer-0 input activations.
// ---------------------------------------------------------------------------
__global__ void __launch_bounds__(256)
split_kernel(const float* __restrict__ src,
             bf16_t* __restrict__ hi,
             bf16_t* __restrict__ lo,
             int n) {
    int i = blockIdx.x * blockDim.x + threadIdx.x;
    if (i < n) {
        float x = src[i];
        bf16_t h = (bf16_t)x;
        hi[i] = h;
        lo[i] = (bf16_t)(x - (float)h);
    }
}

// ---------------------------------------------------------------------------
// Fused GRU layer (h_prev == 0). Per wave: 4 M-tiles (64 rows) x 16 N-cols x
// 3 gates; K-loop in steps of 32; 9 bf16 WMMAs per M-tile per k-step.
// A operand arrives pre-split (Xhi/Xlo bf16 planes) -> no conversions in loop.
// MODE 0: write h1 as bf16 hi/lo planes (feeds layer 1's A operand directly).
// MODE 1: write out[l, b, h] = h2 * mask[b, l]  (fp32, transposed).
// ---------------------------------------------------------------------------
template <int MODE>
__global__ void __launch_bounds__(256)
gru_layer_kernel(const bf16_t* __restrict__ Xhi,   // [M, K] bf16 hi
                 const bf16_t* __restrict__ Xlo,   // [M, K] bf16 lo
                 const bf16_t* __restrict__ Whi,   // [3H, K] bf16 hi
                 const bf16_t* __restrict__ Wlo,   // [3H, K] bf16 lo
                 const float*  __restrict__ b_ih,  // [3H]
                 const float*  __restrict__ b_hh,  // [3H]
                 const int*    __restrict__ mask,  // [B, L] (MODE 1 only)
                 bf16_t*       __restrict__ OutHi, // MODE 0: [M, H] bf16 hi
                 bf16_t*       __restrict__ OutLo, // MODE 0: [M, H] bf16 lo
                 float*        __restrict__ OutF)  // MODE 1: [L, B, H] fp32
{
    const int lane = threadIdx.x & 31;
    const int wave = threadIdx.x >> 5;
    const int l16  = lane & 15;          // C/D: this lane's column index
    const int lhi  = lane >> 4;          // lane half selects K sub-range
    const int h0   = blockIdx.x * 128 + wave * 16;   // this wave's hidden cols
    const int m0   = blockIdx.y * 64;                // workgroup row base
    const int ncol = h0 + l16;                       // this lane's hidden col

    v8f acc[4][3];
#pragma unroll
    for (int mt = 0; mt < 4; ++mt)
#pragma unroll
        for (int g = 0; g < 3; ++g)
#pragma unroll
            for (int i = 0; i < 8; ++i) acc[mt][g][i] = 0.0f;

    // Base pointers; all per-gate / per-M-tile deltas become immediate offsets
    // (gate stride 2 MB, M-tile stride 32 KB -- both < 24-bit signed imm).
    // A k order used here (same as B; not the ISA table order): per lane, 16 contiguous K at lhi*16.
    // B k order used here (same as A): per lane, 16 contiguous K at lhi*16.
    const bf16_t* pAhi = Xhi + (size_t)(m0 + l16) * GRU_K + lhi * 16;
    const bf16_t* pAlo = Xlo + (size_t)(m0 + l16) * GRU_K + lhi * 16;
    const bf16_t* pBhi = Whi + (size_t)ncol * GRU_K + lhi * 16;
    const bf16_t* pBlo = Wlo + (size_t)ncol * GRU_K + lhi * 16;

    for (int kk = 0; kk < GRU_K / 32; ++kk) {
        const int kbase = kk * 32;

        // ---- B fragments: 3 gates x (hi, lo)
        v16bf Bhi[3], Blo[3];
#pragma unroll
        for (int g = 0; g < 3; ++g) {
            const size_t go = (size_t)g * (GRU_H * GRU_K) + kbase;
            v8bf bh0 = *reinterpret_cast<const v8bf*>(pBhi + go);
            v8bf bh1 = *reinterpret_cast<const v8bf*>(pBhi + go + 8);
            v8bf bl0 = *reinterpret_cast<const v8bf*>(pBlo + go);
            v8bf bl1 = *reinterpret_cast<const v8bf*>(pBlo + go + 8);
#pragma unroll
            for (int j = 0; j < 8; ++j) {
                Bhi[g][j]     = bh0[j];
                Bhi[g][j + 8] = bh1[j];
                Blo[g][j]     = bl0[j];
                Blo[g][j + 8] = bl1[j];
            }
        }

        // ---- A fragments per M-tile (already bf16 hi/lo: pure loads)
#pragma unroll
        for (int mt = 0; mt < 4; ++mt) {
            const size_t ao = (size_t)mt * (16 * GRU_K) + kbase;
            v8bf ah0 = *reinterpret_cast<const v8bf*>(pAhi + ao);
            v8bf ah1 = *reinterpret_cast<const v8bf*>(pAhi + ao + 8);
            v8bf al0 = *reinterpret_cast<const v8bf*>(pAlo + ao);
            v8bf al1 = *reinterpret_cast<const v8bf*>(pAlo + ao + 8);
            v16bf Ahi, Alo;
#pragma unroll
            for (int j = 0; j < 8; ++j) {
                Ahi[j]     = ah0[j];
                Ahi[j + 8] = ah1[j];
                Alo[j]     = al0[j];
                Alo[j + 8] = al1[j];
            }
#pragma unroll
            for (int g = 0; g < 3; ++g) {
                acc[mt][g] = __builtin_amdgcn_wmma_f32_16x16x32_bf16(
                    false, Ahi, false, Bhi[g], (short)0, acc[mt][g], false, false);
                acc[mt][g] = __builtin_amdgcn_wmma_f32_16x16x32_bf16(
                    false, Ahi, false, Blo[g], (short)0, acc[mt][g], false, false);
                acc[mt][g] = __builtin_amdgcn_wmma_f32_16x16x32_bf16(
                    false, Alo, false, Bhi[g], (short)0, acc[mt][g], false, false);
            }
        }
    }

    // ---- Fused gate epilogue. C/D layout: lane holds column ncol, rows
    // m0 + mt*16 + i + 8*lhi for vgpr element i.
    const float bihR = b_ih[ncol];
    const float bihZ = b_ih[ncol + GRU_H];
    const float bihN = b_ih[ncol + 2 * GRU_H];
    const float bhhR = b_hh[ncol];
    const float bhhZ = b_hh[ncol + GRU_H];
    const float bhhN = b_hh[ncol + 2 * GRU_H];

#pragma unroll
    for (int mt = 0; mt < 4; ++mt) {
#pragma unroll
        for (int i = 0; i < 8; ++i) {
            const int m = m0 + mt * 16 + i + lhi * 8;
            const float r = sigmoid_f(acc[mt][0][i] + bihR + bhhR);
            const float z = sigmoid_f(acc[mt][1][i] + bihZ + bhhZ);
            const float n = tanhf(acc[mt][2][i] + bihN + r * bhhN);
            const float h = (1.0f - z) * n;
            if (MODE == 0) {
                // Store h1 pre-split for layer 1's A operand.
                const size_t o = (size_t)m * GRU_H + ncol;
                bf16_t hh = (bf16_t)h;
                OutHi[o] = hh;
                OutLo[o] = (bf16_t)(h - (float)hh);
            } else {
                const int b = m >> 11;        // m / L  (L = 2048)
                const int l = m & 2047;       // m % L
                const float mv = (float)mask[m];
                OutF[((size_t)l * GRU_B + b) * GRU_H + ncol] = h * mv;
            }
        }
    }
}

// ---------------------------------------------------------------------------
// Inputs (setup_inputs order):
//  0 inputs (B,L,NIN) f32       1 inputs_mask (B,L) i32
//  2 W_ih0 (3H,NIN) f32  3 W_hh0  4 b_ih0 (3H)  5 b_hh0 (3H)
//  6 W_ih1 (3H,H)   f32  7 W_hh1  8 b_ih1 (3H)  9 b_hh1 (3H)
// Output: (L, B, H) f32.
// ---------------------------------------------------------------------------
extern "C" void kernel_launch(void* const* d_in, const int* in_sizes, int n_in,
                              void* d_out, int out_size, void* d_ws, size_t ws_size,
                              hipStream_t stream) {
    (void)in_sizes; (void)n_in; (void)out_size; (void)ws_size;

    const float* x     = (const float*)d_in[0];
    const int*   mask  = (const int*)d_in[1];
    const float* W0    = (const float*)d_in[2];
    const float* b_ih0 = (const float*)d_in[4];
    const float* b_hh0 = (const float*)d_in[5];
    const float* W1    = (const float*)d_in[6];
    const float* b_ih1 = (const float*)d_in[8];
    const float* b_hh1 = (const float*)d_in[9];
    float* out = (float*)d_out;

    // Workspace layout (bytes):
    //   W0hi/W0lo/W1hi/W1lo : 4 x 6,291,456      (3H*K bf16)
    //   Xhi/Xlo             : 2 x 67,108,864     (M*K bf16)
    //   H1hi/H1lo           : 2 x 67,108,864     (M*H bf16)
    const size_t WELEMS = (size_t)3 * GRU_H * GRU_K;   // 3,145,728
    const size_t WBYTES = WELEMS * sizeof(bf16_t);     // 6,291,456
    const size_t XELEMS = (size_t)GRU_M * GRU_K;       // 33,554,432
    const size_t XBYTES = XELEMS * sizeof(bf16_t);     // 67,108,864
    char* ws = (char*)d_ws;
    bf16_t* W0hi = (bf16_t*)(ws);
    bf16_t* W0lo = (bf16_t*)(ws + WBYTES);
    bf16_t* W1hi = (bf16_t*)(ws + 2 * WBYTES);
    bf16_t* W1lo = (bf16_t*)(ws + 3 * WBYTES);
    bf16_t* Xhi  = (bf16_t*)(ws + 4 * WBYTES);
    bf16_t* Xlo  = (bf16_t*)(ws + 4 * WBYTES + XBYTES);
    bf16_t* H1hi = (bf16_t*)(ws + 4 * WBYTES + 2 * XBYTES);
    bf16_t* H1lo = (bf16_t*)(ws + 4 * WBYTES + 3 * XBYTES);

    // 1) hi/lo pre-splits (weights ~25 MB, activations ~268 MB traffic ~12 us)
    {
        const int nw = (int)WELEMS;
        split_kernel<<<(nw + 255) / 256, 256, 0, stream>>>(W0, W0hi, W0lo, nw);
        split_kernel<<<(nw + 255) / 256, 256, 0, stream>>>(W1, W1hi, W1lo, nw);
        const int nx = (int)XELEMS;
        split_kernel<<<(nx + 255) / 256, 256, 0, stream>>>(x, Xhi, Xlo, nx);
    }

    // 2) Layer 0: h1 = gru(x, W0, b0) -> pre-split bf16 planes in workspace.
    // 3) Layer 1: out = transpose(mask * gru(h1, W1, b1)).
    dim3 grid(GRU_H / 128, GRU_M / 64);   // (8, 512)
    dim3 block(256);
    gru_layer_kernel<0><<<grid, block, 0, stream>>>(
        Xhi, Xlo, W0hi, W0lo, b_ih0, b_hh0, nullptr, H1hi, H1lo, nullptr);
    gru_layer_kernel<1><<<grid, block, 0, stream>>>(
        H1hi, H1lo, W1hi, W1lo, b_ih1, b_hh1, mask, nullptr, nullptr, out);
}
